// QLSTMCell_65481071406668
// MI455X (gfx1250) — matrix operand order repaired and checked by reading, not compiled, not run on hardware
//
#include <hip/hip_runtime.h>
#include <math.h>

typedef __attribute__((ext_vector_type(16))) _Float16 v16h;
typedef __attribute__((ext_vector_type(8)))  float    v8f;
typedef __attribute__((ext_vector_type(4)))  float    vf4;   // true vector type: NT-load/store legal

namespace {
constexpr int kBatch = 8192;
constexpr int kIn    = 2048;
constexpr int kHid   = 2048;
constexpr int kFan   = 4096;   // kIn + kHid
}

// A fragment: 16x32 f16, M rows x K. Lane<16: row=lane, K k0+[0..7] and k0+[16..23].
// Lane>=16: row=lane-16, K k0+[8..15] and k0+[24..31].
// x/hx are streamed exactly once -> non-temporal loads (don't pollute cache vs. hot weights).
__device__ __forceinline__ v16h load_a_frag(const float* __restrict__ x,
                                            const float* __restrict__ hx,
                                            int row, int k0, int lane) {
  int kb = k0 + ((lane & 16) ? 8 : 0);
  const float* base = (k0 < kIn) ? (x + (size_t)row * kIn + kb)
                                 : (hx + (size_t)row * kHid + (kb - kIn));
  vf4 f0 = __builtin_nontemporal_load((const vf4*)base + 0);
  vf4 f1 = __builtin_nontemporal_load((const vf4*)base + 1);
  vf4 f2 = __builtin_nontemporal_load((const vf4*)base + 4);   // base + 16 floats
  vf4 f3 = __builtin_nontemporal_load((const vf4*)base + 5);   // base + 20 floats
  v16h a;
  a[0]=(_Float16)f0.x; a[1]=(_Float16)f0.y; a[2]=(_Float16)f0.z; a[3]=(_Float16)f0.w;
  a[4]=(_Float16)f1.x; a[5]=(_Float16)f1.y; a[6]=(_Float16)f1.z; a[7]=(_Float16)f1.w;
  a[8]=(_Float16)f2.x; a[9]=(_Float16)f2.y; a[10]=(_Float16)f2.z; a[11]=(_Float16)f2.w;
  a[12]=(_Float16)f3.x; a[13]=(_Float16)f3.y; a[14]=(_Float16)f3.z; a[15]=(_Float16)f3.w;
  return a;
}

// B fragment: 32x16 f16, K x N. Column = lane&15. Lanes<16 hold K k0+[0..7],[16..23],
// lanes>=16 hold K k0+[8..15],[24..31]. Weights are tiny (512KB) and reused by every
// wave -> regular (cached) loads.
__device__ __forceinline__ v16h load_b_frag(const float* __restrict__ Wgate,
                                            int wire, int k0, int lane) {
  int k = k0 + ((lane & 16) ? 8 : 0);
  const float* p = Wgate + (size_t)wire * kFan + k;
  vf4 f0 = ((const vf4*)p)[0];
  vf4 f1 = ((const vf4*)p)[1];
  vf4 f2 = ((const vf4*)p)[4];
  vf4 f3 = ((const vf4*)p)[5];
  v16h b;
  b[0]=(_Float16)f0.x; b[1]=(_Float16)f0.y; b[2]=(_Float16)f0.z; b[3]=(_Float16)f0.w;
  b[4]=(_Float16)f1.x; b[5]=(_Float16)f1.y; b[6]=(_Float16)f1.z; b[7]=(_Float16)f1.w;
  b[8]=(_Float16)f2.x; b[9]=(_Float16)f2.y; b[10]=(_Float16)f2.z; b[11]=(_Float16)f2.w;
  b[12]=(_Float16)f3.x; b[13]=(_Float16)f3.y; b[14]=(_Float16)f3.z; b[15]=(_Float16)f3.w;
  return b;
}

// Phase 1: C[8192,32] = [x|hx] @ Wall^T  (n: 0-7=f wires, 8-15=i, 16-23=g, 24-31=o),
// then sum_{b,n in gate} cos(C + bias) -> 4 global scalars (atomic f32).
// Block = 256 thr (8 wave32s). Waves 0-3: tile 2*blockIdx, K-chunks 0-3.
// Waves 4-7: tile 2*blockIdx+1. (2048 waves total for deep HBM pipelining.)
__global__ void __launch_bounds__(256) qlstm_gemm_cos(
    const float* __restrict__ x,  const float* __restrict__ hx,
    const float* __restrict__ Wf, const float* __restrict__ Wi,
    const float* __restrict__ Wg, const float* __restrict__ Wo,
    const float* __restrict__ bf, const float* __restrict__ bi,
    const float* __restrict__ bg, const float* __restrict__ bo,
    float* __restrict__ gsum_global)
{
  __shared__ float tileC[2][2][16][16];   // [tileInWG][nHalf][row][col]
  __shared__ float gsum[4];

  const int tid = threadIdx.x;
  for (int i = tid; i < 1024; i += 256) ((float*)tileC)[i] = 0.0f;
  if (tid < 4) gsum[tid] = 0.0f;
  __syncthreads();

  const int lane = tid & 31;
  const int wave = tid >> 5;
  const int tin  = wave >> 2;        // tile within workgroup (0/1)
  const int kc   = wave & 3;         // K chunk (0..3), 1024 each
  const int tile = blockIdx.x * 2 + tin;
  const int rowA = tile * 16 + (lane & 15);
  const int col  = lane & 15;

  // Gate/wire selection per B column (uniform per lane across the K loop)
  const int n0 = col;        // half 0: n in [0,16)
  const int n1 = 16 + col;   // half 1: n in [16,32)
  const int wire0 = n0 & 7, wire1 = n1 & 7;
  const float* Wn0 = (n0 < 8)  ? Wf : Wi;
  const float* Wn1 = (n1 < 24) ? Wg : Wo;

  v8f c0 = {}; v8f c1 = {};
  const int kbeg = kc * 1024, kend = kbeg + 1024;
#pragma unroll 2
  for (int k0 = kbeg; k0 < kend; k0 += 32) {
    v16h a  = load_a_frag(x, hx, rowA, k0, lane);
    v16h b0 = load_b_frag(Wn0, wire0, k0, lane);
    v16h b1 = load_b_frag(Wn1, wire1, k0, lane);
    c0 = __builtin_amdgcn_wmma_f32_16x16x32_f16(false, a, false, b0, (short)0, c0, false, false);
    c1 = __builtin_amdgcn_wmma_f32_16x16x32_f16(false, a, false, b1, (short)0, c1, false, false);
  }

  // Merge K-chunk partial C tiles in LDS (C layout: row = r + 8*(lane>=16), col = lane&15)
  const int rbase = (lane & 16) ? 8 : 0;
#pragma unroll
  for (int r = 0; r < 8; ++r) {
    atomicAdd(&tileC[tin][0][rbase + r][col], c0[r]);
    atomicAdd(&tileC[tin][1][rbase + r][col], c1[r]);
  }
  __syncthreads();

  // bias + cos + reduce: 1024 elements, 4 consecutive per thread (same gate per thread)
  {
    const int idx0 = tid * 4;
    const int h0   = (idx0 >> 8) & 1;
    const int gate = ((h0 << 4) + (idx0 & 15)) >> 3;
    const float* bptr = (gate == 0) ? bf : (gate == 1) ? bi : (gate == 2) ? bg : bo;
    float s = 0.0f;
#pragma unroll
    for (int j = 0; j < 4; ++j) {
      int idx  = idx0 + j;
      int t    = idx >> 9;
      int h    = (idx >> 8) & 1;
      int rrow = (idx >> 4) & 15;
      int ccol = idx & 15;
      int wire = ((h << 4) + ccol) & 7;
      s += cosf(tileC[t][h][rrow][ccol] + bptr[wire]);
    }
    atomicAdd(&gsum[gate], s);
  }
  __syncthreads();
  if (tid < 4) atomicAdd(&gsum_global[tid], gsum[tid]);
}

// Phase 2: scalar gates -> elementwise LSTM update. Pure streaming (201 MB):
// NT loads for cx (read-once), NT stores for both outputs.
__global__ void __launch_bounds__(256) qlstm_elem(
    const float* __restrict__ cx, const float* __restrict__ gs,
    float* __restrict__ outh, float* __restrict__ outc)
{
  const float f  = 1.0f / (1.0f + expf(-gs[0]));
  const float ii = 1.0f / (1.0f + expf(-gs[1]));
  const float g  = tanhf(gs[2]);
  const float o  = 1.0f / (1.0f + expf(-gs[3]));
  const float ig = ii * g;

  const int n4 = (kBatch * kHid) / 4;
  const int stride = gridDim.x * blockDim.x;
  for (int idx = blockIdx.x * blockDim.x + threadIdx.x; idx < n4; idx += stride) {
    vf4 c = __builtin_nontemporal_load((const vf4*)cx + idx);
    vf4 cn, hn;
    cn.x = f * c.x + ig;  cn.y = f * c.y + ig;
    cn.z = f * c.z + ig;  cn.w = f * c.w + ig;
    hn.x = o * tanhf(cn.x); hn.y = o * tanhf(cn.y);
    hn.z = o * tanhf(cn.z); hn.w = o * tanhf(cn.w);
    __builtin_nontemporal_store(cn, (vf4*)outc + idx);
    __builtin_nontemporal_store(hn, (vf4*)outh + idx);
  }
}

extern "C" void kernel_launch(void* const* d_in, const int* in_sizes, int n_in,
                              void* d_out, int out_size, void* d_ws, size_t ws_size,
                              hipStream_t stream) {
  (void)in_sizes; (void)n_in; (void)out_size; (void)ws_size;
  const float* x  = (const float*)d_in[0];
  const float* hx = (const float*)d_in[1];
  const float* cx = (const float*)d_in[2];
  const float* Wf = (const float*)d_in[3];
  const float* bf = (const float*)d_in[4];
  const float* Wi = (const float*)d_in[5];
  const float* bi = (const float*)d_in[6];
  const float* Wg = (const float*)d_in[7];
  const float* bg = (const float*)d_in[8];
  const float* Wo = (const float*)d_in[9];
  const float* bo = (const float*)d_in[10];

  float* gsums = (float*)d_ws;
  float* outh  = (float*)d_out;
  float* outc  = outh + (size_t)kBatch * kHid;

  hipMemsetAsync(gsums, 0, 4 * sizeof(float), stream);
  // 512 M-tiles of 16 rows, 2 tiles per block, 4 K-chunk waves per tile.
  qlstm_gemm_cos<<<256, 256, 0, stream>>>(x, hx, Wf, Wi, Wg, Wo, bf, bi, bg, bo, gsums);
  qlstm_elem<<<4096, 256, 0, stream>>>(cx, gsums, outh, outc);
}
